// TensorTrainFixedRank_37048387896002
// MI455X (gfx1250) — matrix operand order repaired and checked by reading, not compiled, not run on hardware
//
#include <hip/hip_runtime.h>
#include <hip/hip_bf16.h>

typedef __bf16 v8bf  __attribute__((ext_vector_type(8)));
typedef __bf16 v16bf __attribute__((ext_vector_type(16)));
typedef float  v8f   __attribute__((ext_vector_type(8)));

#define B_BATCH 8192
#define D_FEAT  8
#define P_DEG   10
#define R_RANK  512
#define OUT_DIM 64
#define K_TT    (R_RANK * P_DEG)   // 5120

#if defined(__AMDGCN__) && __has_builtin(__builtin_amdgcn_global_load_async_to_lds_b128)
#define USE_ASYNC_LDS 1
// builtin expects v4i pointers: AS1 (global) source, AS3 (LDS) dest
typedef int v4i_async __attribute__((vector_size(16)));
typedef __attribute__((address_space(1))) v4i_async* as1_v4i;
typedef __attribute__((address_space(3))) v4i_async* as3_v4i;
#else
#define USE_ASYNC_LDS 0
#endif

__device__ __forceinline__ unsigned short f2bf(float f) {
    unsigned int u = __float_as_uint(f);
    u += 0x7fffu + ((u >> 16) & 1u);        // round-to-nearest-even
    return (unsigned short)(u >> 16);
}

// ---------- prep: W0 = tanh(G0), f32 [10][512] ----------
__global__ void k_tanh_w0(const float* __restrict__ g0, float* __restrict__ w0) {
    int i = blockIdx.x * blockDim.x + threadIdx.x;
    if (i < P_DEG * R_RANK) w0[i] = tanhf(g0[i]);
}

// ---------- prep: out[n][k] = bf16(tanh(in[k][n])), tiled transpose ----------
__global__ void k_tanh_transpose(const float* __restrict__ in, unsigned short* __restrict__ out,
                                 int K, int N) {
    __shared__ float tile[32][33];
    const int k0 = blockIdx.x * 32, n0 = blockIdx.y * 32;
    const int tx = threadIdx.x, ty = threadIdx.y;
    #pragma unroll
    for (int i = 0; i < 4; ++i) {
        int k = k0 + ty + i * 8;
        int n = n0 + tx;                     // coalesced read (n contiguous)
        tile[tx][ty + i * 8] = tanhf(in[(size_t)k * N + n]);   // tile[n_loc][k_loc]
    }
    __syncthreads();
    #pragma unroll
    for (int i = 0; i < 4; ++i) {
        int n = n0 + ty + i * 8;
        int k = k0 + tx;                     // coalesced write (k contiguous)
        out[(size_t)n * K + k] = f2bf(tile[ty + i * 8][tx]);
    }
}

// ---------- first core: res0[b,r] = sum_j tanh(G0)[j,r] * z[b,0]^j ----------
__global__ void k_core0(const float* __restrict__ z, const float* __restrict__ w0,
                        float* __restrict__ res0) {
    int idx = blockIdx.x * blockDim.x + threadIdx.x;   // B*R threads
    int b = idx >> 9, r = idx & (R_RANK - 1);
    float zb = z[b * D_FEAT + 0];
    float acc = 0.f, pw = 1.f;
    #pragma unroll
    for (int j = 0; j < P_DEG; ++j) { acc += w0[j * R_RANK + r] * pw; pw *= zb; }
    res0[idx] = acc;
}

// ---------- fused TT contraction:  res_out = (res_in (x) phi) @ Wt^T ----------
// Wt is bf16 [Nout][K_TT] (K contiguous).  A[b, i*10+j] = res_in[b,i] * z[b,feat]^j
template<int BM, int BN, int SK, int WAVES_M, int WAVES_N>
__global__ void __launch_bounds__(WAVES_M * WAVES_N * 32)
k_tt_core(const float* __restrict__ z, const float* __restrict__ res_in,
          const unsigned short* __restrict__ Wt, float* __restrict__ res_out,
          int feat, int Nout) {
    constexpr int T       = WAVES_M * WAVES_N * 32;
    constexpr int WMROWS  = BM / WAVES_M;      // rows per wave
    constexpr int WNCOLS  = BN / WAVES_N;      // cols per wave
    constexpr int FM      = WMROWS / 16;
    constexpr int FN      = WNCOLS / 16;
    constexpr int SKP     = SK + 8;            // padded LDS row stride (144B: kills
                                               // the 128B-stride 16-way bank conflict,
                                               // keeps 16B alignment)

    __shared__ __align__(16) unsigned short Als[BM * SKP];   // A tile, bf16
    __shared__ __align__(16) unsigned short Bls[BN * SKP];   // W tile, bf16 (row = out col)
    __shared__ float phis[BM * P_DEG];

    const int tid  = threadIdx.x;
    const int lane = tid & 31;
    const int w    = tid >> 5;
    const int wm   = w % WAVES_M;
    const int wn   = w / WAVES_M;
    const int gm   = blockIdx.y * BM;
    const int gn   = blockIdx.x * BN;

    // per-row polynomial basis phi[b, j] = z[b,feat]^j
    for (int r = tid; r < BM; r += T) {
        float zb = z[(gm + r) * D_FEAT + feat];
        float pw = 1.f;
        #pragma unroll
        for (int j = 0; j < P_DEG; ++j) { phis[r * P_DEG + j] = pw; pw *= zb; }
    }

    v8f acc[FM][FN];
    #pragma unroll
    for (int mi = 0; mi < FM; ++mi)
        #pragma unroll
        for (int ni = 0; ni < FN; ++ni)
            #pragma unroll
            for (int e = 0; e < 8; ++e) acc[mi][ni][e] = 0.f;

    __syncthreads();   // phis ready

    for (int slab = 0; slab < K_TT / SK; ++slab) {
        const int k0 = slab * SK;

        // ---- weight tile: global -> LDS (async DMA path on CDNA5) ----
        for (int p = tid; p < BN * (SK / 8); p += T) {
            int r = p / (SK / 8), c = p % (SK / 8);
            const unsigned short* gsrc = Wt + (size_t)(gn + r) * K_TT + k0 + c * 8;
            unsigned short*       ldst = Bls + r * SKP + c * 8;
#if USE_ASYNC_LDS
            __builtin_amdgcn_global_load_async_to_lds_b128(
                (as1_v4i)(uintptr_t)(const void*)gsrc,
                (as3_v4i)(unsigned int)(uintptr_t)(void*)ldst,
                0, 0);
#else
            *reinterpret_cast<uint4*>(ldst) = *reinterpret_cast<const uint4*>(gsrc);
#endif
        }

        // ---- generate A tile while async copy is in flight ----
        // bf16 pairs packed into dwords (k even -> k,k+1 share the same i)
        for (int p = tid; p < BM * (SK / 2); p += T) {
            int r  = p / (SK / 2), kp = p % (SK / 2);
            int k  = k0 + 2 * kp;
            int i  = k / P_DEG;
            int j  = k - i * P_DEG;
            float rv = res_in[(gm + r) * R_RANK + i];
            unsigned int lo = f2bf(rv * phis[r * P_DEG + j]);
            unsigned int hi = f2bf(rv * phis[r * P_DEG + j + 1]);
            reinterpret_cast<unsigned int*>(Als + r * SKP)[kp] = lo | (hi << 16);
        }

#if USE_ASYNC_LDS
#if __has_builtin(__builtin_amdgcn_s_wait_asynccnt)
        __builtin_amdgcn_s_wait_asynccnt(0);
#else
        asm volatile("s_wait_asynccnt 0x0" ::: "memory");
#endif
#endif
        __syncthreads();

        #pragma unroll
        for (int kk = 0; kk < SK / 32; ++kk) {
            // A fragments (ISA 7.12.2, 16-bit A 16x32): lane<16 -> K {0..7,16..23}, lane>=16 -> K {8..15,24..31}
            v16bf af[FM];
            const int kbA = kk * 32 + ((lane >> 4) ? 8 : 0);
            #pragma unroll
            for (int mi = 0; mi < FM; ++mi) {
                int arow = wm * WMROWS + mi * 16 + (lane & 15);
                v8bf lo = *reinterpret_cast<const v8bf*>(Als + arow * SKP + kbA);
                v8bf hi = *reinterpret_cast<const v8bf*>(Als + arow * SKP + kbA + 16);
                af[mi] = __builtin_shufflevector(lo, hi, 0,1,2,3,4,5,6,7,8,9,10,11,12,13,14,15);
            }
            // B fragments: lane<16 -> K {0..7,16..23} (col=lane), lane>=16 -> K {8..15,24..31} (col=lane-16)
            const int kbB = kk * 32 + ((lane >> 4) ? 8 : 0);
            #pragma unroll
            for (int ni = 0; ni < FN; ++ni) {
                int bcol = wn * WNCOLS + ni * 16 + (lane & 15);
                v8bf lo = *reinterpret_cast<const v8bf*>(Bls + bcol * SKP + kbB);
                v8bf hi = *reinterpret_cast<const v8bf*>(Bls + bcol * SKP + kbB + 16);
                v16bf bf = __builtin_shufflevector(lo, hi, 0,1,2,3,4,5,6,7,8,9,10,11,12,13,14,15);
                #pragma unroll
                for (int mi = 0; mi < FM; ++mi)
                    acc[mi][ni] = __builtin_amdgcn_wmma_f32_16x16x32_bf16(
                        false, af[mi], false, bf, (short)0, acc[mi][ni], false, false);
            }
        }
        __syncthreads();
    }

    // store: C/D layout — VGPR e: lane<16 -> M=e, N=lane; lane>=16 -> M=e+8, N=lane-16
    #pragma unroll
    for (int mi = 0; mi < FM; ++mi)
        #pragma unroll
        for (int ni = 0; ni < FN; ++ni) {
            int col   = gn + wn * WNCOLS + ni * 16 + (lane & 15);
            int rbase = gm + wm * WMROWS + mi * 16 + ((lane >> 4) << 3);
            #pragma unroll
            for (int e = 0; e < 8; ++e)
                res_out[(size_t)(rbase + e) * Nout + col] = acc[mi][ni][e];
        }
}

extern "C" void kernel_launch(void* const* d_in, const int* in_sizes, int n_in,
                              void* d_out, int out_size, void* d_ws, size_t ws_size,
                              hipStream_t stream) {
    const float* z      = (const float*)d_in[0];   // [8192][8]
    const float* G0     = (const float*)d_in[1];   // [10][512]
    const float* G_mid  = (const float*)d_in[2];   // [6][512][10][512]
    const float* G_last = (const float*)d_in[3];   // [512][10][64]
    // d_in[4] = t (unused)

    char* ws = (char*)d_ws;
    size_t off = 0;
    float* w0 = (float*)(ws + off);            off += (size_t)P_DEG * R_RANK * 4;
    off = (off + 255) & ~(size_t)255;
    unsigned short* wmid  = (unsigned short*)(ws + off); off += (size_t)6 * R_RANK * K_TT * 2;
    unsigned short* wlast = (unsigned short*)(ws + off); off += (size_t)OUT_DIM * K_TT * 2;
    off = (off + 255) & ~(size_t)255;
    float* resA = (float*)(ws + off);          off += (size_t)B_BATCH * R_RANK * 4;
    float* resB = (float*)(ws + off);          off += (size_t)B_BATCH * R_RANK * 4;

    // weight prep: tanh (+ bf16 transpose for WMMA B operand)
    k_tanh_w0<<<(P_DEG * R_RANK + 255) / 256, 256, 0, stream>>>(G0, w0);
    for (int c = 0; c < 6; ++c)
        k_tanh_transpose<<<dim3(K_TT / 32, R_RANK / 32), dim3(32, 8), 0, stream>>>(
            G_mid + (size_t)c * K_TT * R_RANK, wmid + (size_t)c * R_RANK * K_TT,
            K_TT, R_RANK);
    k_tanh_transpose<<<dim3(K_TT / 32, OUT_DIM / 32), dim3(32, 8), 0, stream>>>(
        G_last, wlast, K_TT, OUT_DIM);

    // first core
    k_core0<<<(B_BATCH * R_RANK) / 256, 256, 0, stream>>>(z, w0, resA);

    // middle cores: fused rank-1-expansion GEMMs
    float* cur = resA; float* nxt = resB;
    for (int c = 0; c < 6; ++c) {
        k_tt_core<128, 128, 64, 4, 2><<<dim3(R_RANK / 128, B_BATCH / 128), 256, 0, stream>>>(
            z, cur, wmid + (size_t)c * R_RANK * K_TT, nxt, c + 1, R_RANK);
        float* t = cur; cur = nxt; nxt = t;
    }
    // last core -> d_out (f32 [8192][64])
    k_tt_core<128, 64, 64, 4, 2><<<dim3(OUT_DIM / 64, B_BATCH / 128), 256, 0, stream>>>(
        z, cur, wlast, (float*)d_out, D_FEAT - 1, OUT_DIM);
}
